// MoELinear_77764677861843
// MI455X (gfx1250) — matrix operand order repaired and checked by reading, not compiled, not run on hardware
//
#include <hip/hip_runtime.h>
#include <hip/hip_bf16.h>
#include <math.h>

#define N_EXPERTS 28
#define RANK 8
#define LORA_SCALING 2.0f   // lora_alpha / r = 16 / 8

typedef __attribute__((ext_vector_type(16))) __bf16 v16bf;
typedef __attribute__((ext_vector_type(8)))  __bf16 v8bf;
typedef __attribute__((ext_vector_type(4)))  __bf16 v4bf;
typedef __attribute__((ext_vector_type(8)))  float  v8f;

// ---------------------------------------------------------------------------
// Kernel 1: gating (top-2 + softmax) and LoRA-A projection for the selected
// experts. One wave32 per token. Writes idx[N][2] and h[N][16] (pre-scaled by
// gate prob * SCALING) into workspace.
// ---------------------------------------------------------------------------
__global__ __launch_bounds__(256) void moe_gate_h_kernel(
    const float* __restrict__ x,       // [N][d_in]
    const float* __restrict__ gate_W,  // [28][d_in]
    const float* __restrict__ lora_A,  // [28][RANK][d_in]
    float* __restrict__ h_out,         // [N][2*RANK]
    int*   __restrict__ idx_out,       // [N][2]
    int n_tok, int d_in)
{
  const int lane = threadIdx.x & 31;
  const int wave = threadIdx.x >> 5;
  const int n = blockIdx.x * 8 + wave;
  if (n >= n_tok) return;
  const float* xr = x + (size_t)n * d_in;

  // ---- gate logits: 28 dot products, lanes stride over d_in ----
  float acc[N_EXPERTS];
#pragma unroll
  for (int e = 0; e < N_EXPERTS; ++e) acc[e] = 0.f;
  for (int d = lane; d < d_in; d += 32) {
    float xv = xr[d];
#pragma unroll
    for (int e = 0; e < N_EXPERTS; ++e)
      acc[e] = fmaf(xv, gate_W[(size_t)e * d_in + d], acc[e]);
  }
#pragma unroll
  for (int e = 0; e < N_EXPERTS; ++e) {
#pragma unroll
    for (int off = 16; off > 0; off >>= 1)
      acc[e] += __shfl_xor(acc[e], off, 32);
  }

  // ---- top-2 (ties keep lowest index, matching jax top_k) ----
  int i0 = 0; float v0 = acc[0];
#pragma unroll
  for (int e = 1; e < N_EXPERTS; ++e)
    if (acc[e] > v0) { v0 = acc[e]; i0 = e; }
  int i1 = (i0 == 0) ? 1 : 0; float v1 = acc[i1];
#pragma unroll
  for (int e = 0; e < N_EXPERTS; ++e)
    if (e != i0 && acc[e] > v1) { v1 = acc[e]; i1 = e; }

  // softmax over the two selected logits (T = 1)
  float p0 = 1.f / (1.f + __expf(v1 - v0));
  float p1 = 1.f - p0;
  float w0 = LORA_SCALING * p0, w1 = LORA_SCALING * p1;

  // ---- LoRA-A projection for the two selected experts ----
  const float* A0 = lora_A + (size_t)i0 * RANK * d_in;
  const float* A1 = lora_A + (size_t)i1 * RANK * d_in;
  float ha[2 * RANK];
#pragma unroll
  for (int r = 0; r < 2 * RANK; ++r) ha[r] = 0.f;
  for (int d = lane; d < d_in; d += 32) {
    float xv = xr[d];
#pragma unroll
    for (int r = 0; r < RANK; ++r) {
      ha[r]        = fmaf(xv, A0[(size_t)r * d_in + d], ha[r]);
      ha[RANK + r] = fmaf(xv, A1[(size_t)r * d_in + d], ha[RANK + r]);
    }
  }
#pragma unroll
  for (int r = 0; r < 2 * RANK; ++r) {
#pragma unroll
    for (int off = 16; off > 0; off >>= 1)
      ha[r] += __shfl_xor(ha[r], off, 32);
  }
  if (lane == 0) {
    idx_out[2 * n]     = i0;
    idx_out[2 * n + 1] = i1;
    float* hp = h_out + (size_t)n * (2 * RANK);
#pragma unroll
    for (int r = 0; r < RANK; ++r) {
      hp[r]        = ha[r]        * w0;
      hp[RANK + r] = ha[RANK + r] * w1;
    }
  }
}

// ---------------------------------------------------------------------------
// Kernel 2: fused base GEMM (bf16 WMMA, f32 accumulate) + LoRA-B epilogue.
// Block tile 128(M) x 256(N) x 32(K); 8 wave32s in a 2x4 grid, each wave owns
// a 64x64 sub-tile = 16 v_wmma_f32_16x16x32_bf16 per K-step (1.0 ds_load_b128
// per WMMA, 4x fragment reuse).
// ---------------------------------------------------------------------------
#define BM 128
#define BN 256
#define BK 32
#define LDSK (BK + 8)   // 40 bf16 = 80B row stride: 16B aligned, bank-conflict free

// 256 threads cooperatively load a ROWS x 32 f32 tile, convert to bf16 -> LDS
template <int ROWS>
__device__ __forceinline__ void tile_load_cvt(const float* __restrict__ src,
                                              int src_stride,
                                              __bf16* __restrict__ dst, int t)
{
#pragma unroll
  for (int rr = 0; rr < ROWS; rr += 128) {
    const int r  = rr + (t >> 1);   // 0..ROWS-1
    const int c0 = (t & 1) * 16;    // 0 or 16
    const float4* p = (const float4*)(src + (size_t)r * src_stride + c0);
    __bf16* q = dst + r * LDSK + c0;
#pragma unroll
    for (int i = 0; i < 4; ++i) {
      float4 v = p[i];
      v4bf b;
      b.x = (__bf16)v.x; b.y = (__bf16)v.y; b.z = (__bf16)v.z; b.w = (__bf16)v.w;
      *(v4bf*)(q + i * 4) = b;      // ds_store_b64
    }
  }
}

__global__ __launch_bounds__(256) void moe_base_lora_gemm_kernel(
    const float* __restrict__ x,       // [N][d_in]
    const float* __restrict__ W,       // [d_out][d_in]
    const float* __restrict__ lora_B,  // [28][d_out][RANK]
    const float* __restrict__ h,       // [N][2*RANK] pre-scaled
    const int*   __restrict__ idx,     // [N][2]
    float* __restrict__ out,           // [N][d_out]
    int d_in, int d_out)
{
  __shared__ __bf16 sA[2][BM * LDSK];   // 2 x 10 KB
  __shared__ __bf16 sB[2][BN * LDSK];   // 2 x 20 KB

  const int t    = threadIdx.x;
  const int lane = t & 31;
  const int wv   = t >> 5;      // 8 waves
  const int wm   = wv >> 2;     // 0..1 : 64-row slab
  const int wn   = wv & 3;      // 0..3 : 64-col slab
  const int lh   = lane >> 4;   // half-wave select (K split per ISA layout)
  const int l16  = lane & 15;

  const int m0 = blockIdx.y * BM;
  const int n0 = blockIdx.x * BN;

  const float* Abase = x + (size_t)m0 * d_in;
  const float* Bbase = W + (size_t)n0 * d_in;

  v8f acc[4][4] = {};   // 128 f32 accumulators / lane = 64x64 tile per wave

  const int kTiles = d_in / BK;
  tile_load_cvt<BM>(Abase, d_in, sA[0], t);
  tile_load_cvt<BN>(Bbase, d_in, sB[0], t);
  __syncthreads();

  for (int kt = 0; kt < kTiles; ++kt) {
    const int cur = kt & 1;
    if (kt + 1 < kTiles) {
      tile_load_cvt<BM>(Abase + (size_t)(kt + 1) * BK, d_in, sA[1 - cur], t);
      tile_load_cvt<BN>(Bbase + (size_t)(kt + 1) * BK, d_in, sB[1 - cur], t);
      if (kt + 2 < kTiles) {
        // pull the K+2 tiles toward L2 (global_prefetch_b8)
        __builtin_prefetch(Abase + (size_t)(t >> 1) * d_in + (size_t)(kt + 2) * BK, 0, 1);
        __builtin_prefetch(Bbase + (size_t)(t >> 1) * d_in + (size_t)(kt + 2) * BK, 0, 1);
        __builtin_prefetch(Bbase + (size_t)(128 + (t >> 1)) * d_in + (size_t)(kt + 2) * BK, 0, 1);
      }
    }

    const __bf16* As = sA[cur];
    const __bf16* Bs = sB[cur];

    // A fragment (16x32 bf16): lane m<16 -> row m, K lh*8..+8 and 16+lh*8..+8
    v16bf af[4];
#pragma unroll
    for (int mi = 0; mi < 4; ++mi) {
      const __bf16* pa = As + (wm * 64 + mi * 16 + l16) * LDSK + lh * 8;
      v8bf lo = *(const v8bf*)pa;          // ds_load_b128
      v8bf hi = *(const v8bf*)(pa + 16);   // ds_load_b128
      af[mi] = __builtin_shufflevector(lo, hi, 0,1,2,3,4,5,6,7,8,9,10,11,12,13,14,15);
    }
    // B fragment (32x16 bf16): lane n holds column n, K lh*8..+8 and 16+lh*8..+8
    v16bf bfr[4];
#pragma unroll
    for (int nj = 0; nj < 4; ++nj) {
      const __bf16* pb = Bs + (wn * 64 + nj * 16 + l16) * LDSK + lh * 8;
      v8bf lo = *(const v8bf*)pb;
      v8bf hi = *(const v8bf*)(pb + 16);
      bfr[nj] = __builtin_shufflevector(lo, hi, 0,1,2,3,4,5,6,7,8,9,10,11,12,13,14,15);
    }

#pragma unroll
    for (int mi = 0; mi < 4; ++mi)
#pragma unroll
      for (int nj = 0; nj < 4; ++nj)
        acc[mi][nj] = __builtin_amdgcn_wmma_f32_16x16x32_bf16(
            false, af[mi], false, bfr[nj], (short)0, acc[mi][nj], false, false);

    __syncthreads();
  }

  // ---- epilogue: out = base + sum_k h[m,k,:]·lora_B[e_k,o,:] ----
  const int mW = m0 + wm * 64;
  const int nW = n0 + wn * 64;
#pragma unroll
  for (int mi = 0; mi < 4; ++mi) {
#pragma unroll
    for (int r = 0; r < 8; ++r) {
      // C/D layout: lanes 0-15 -> M=r, lanes 16-31 -> M=8+r
      const int m  = mW + mi * 16 + lh * 8 + r;
      const int e0 = idx[2 * m];
      const int e1 = idx[2 * m + 1];
      const float* hm = h + (size_t)m * (2 * RANK);
#pragma unroll
      for (int nj = 0; nj < 4; ++nj) {
        const int o = nW + nj * 16 + l16;
        const float* B0 = lora_B + ((size_t)e0 * d_out + o) * RANK;
        const float* B1 = lora_B + ((size_t)e1 * d_out + o) * RANK;
        float lor = 0.f;
#pragma unroll
        for (int j = 0; j < RANK; ++j)
          lor = fmaf(hm[j], B0[j], fmaf(hm[RANK + j], B1[j], lor));
        out[(size_t)m * d_out + o] = acc[mi][nj][r] + lor;
      }
    }
  }
}

// ---------------------------------------------------------------------------
extern "C" void kernel_launch(void* const* d_in, const int* in_sizes, int n_in,
                              void* d_out, int out_size, void* d_ws, size_t ws_size,
                              hipStream_t stream)
{
  const float* x      = (const float*)d_in[0];
  const float* base_W = (const float*)d_in[1];
  const float* gate_W = (const float*)d_in[2];
  const float* lora_A = (const float*)d_in[3];
  const float* lora_B = (const float*)d_in[4];
  float* out = (float*)d_out;

  const int d_in_dim  = in_sizes[2] / N_EXPERTS;   // 4096
  const int n_tok     = in_sizes[0] / d_in_dim;    // 8192
  const int d_out_dim = in_sizes[1] / d_in_dim;    // 4096

  // workspace: idx[N][2] ints, then h[N][16] floats (~576 KB total)
  int*   idx = (int*)d_ws;
  float* h   = (float*)((char*)d_ws + (size_t)n_tok * 2 * sizeof(int));

  moe_gate_h_kernel<<<(n_tok + 7) / 8, 256, 0, stream>>>(
      x, gate_W, lora_A, h, idx, n_tok, d_in_dim);

  dim3 grid(d_out_dim / BN, n_tok / BM);
  moe_base_lora_gemm_kernel<<<grid, 256, 0, stream>>>(
      x, base_W, lora_B, h, idx, out, d_in_dim, d_out_dim);
}
